// GaussianAdaBlurLayer10_inverse_1_out_45535243272819
// MI455X (gfx1250) — matrix operand order repaired and checked by reading, not compiled, not run on hardware
//
#include <hip/hip_runtime.h>
#include <math.h>

// ---------------- types ----------------
typedef __bf16 bf16;
typedef __attribute__((ext_vector_type(16))) __bf16 v16bf;
typedef __attribute__((ext_vector_type(8)))  __bf16 v8bf;
typedef __attribute__((ext_vector_type(8)))  float  v8f;
typedef __attribute__((ext_vector_type(4)))  unsigned int v4u;

#define HH 258
#define WW 258
#define PP (258*258)          // 66564 pixels per plane
#define NN 8
#define MT64 1041             // ceil(PP/64)
#define PB256 261             // ceil(PP/256)

// padded concat buffer: 136 channels (132 used), 262 rows x 260 cols, halo zero
#define PADW 260
#define PADH 262
#define PPAD (PADH*PADW)      // 68120
#define CCH  136

// padded x for bank conv / dyn conv / ax (halo 10*dil, dil clamped to [1,8])
#define HG 427                // 258 + 21*8 + 1
#define WG 418                // 258 + 20*8
// padded x for predictor conv1 (halo 2, 4th channel zero)
#define H1 263
#define W1 262

#define WMMA_BF16(A,B,C) \
  __builtin_amdgcn_wmma_f32_16x16x32_bf16(false,(A),false,(B),(short)0,(C),false,false)

// Fragment-order mapping (CDNA5 ISA 7.12.2, wave32):
//  A slot (msub, lane, h):  m = msub*16 + (lane&15), k = 8*(lane>>4) + h + (h<8?0:8)
//  B slot (nb, lane, h):    n = nb*16 + (lane&15),   k = 8*(lane>>4) + h + (h<8?0:8)
// Staging thread tid packs halfs f = tid*8 + j  ->  msub=tid>>6, lane=(tid>>1)&31,
//  h = (tid&1)*8 + j  =>  k = 8*sel + j + (tid&1)*16.

// ---------------- weight prep: fragment-ready global B matrices ----------------
// BBf[g][14][2][32][16]  bank    (K 441->448 pad, N 22->32 pad)
// B1f[12][8][32][16]     pred_w1 (K 363->384, N=128)
// BFf[38][6][32][16]     fuse_w  (K 1188->1216, N 66->96)
// tw[66][9]              tconv_w summed over its 3 identical input channels
__global__ void k_prep(const float* __restrict__ bank, const float* __restrict__ pw1,
                       const float* __restrict__ fw,   const float* __restrict__ tcw,
                       bf16* BBf, bf16* B1f, bf16* BFf, float* tw) {
  int t = blockIdx.x * blockDim.x + threadIdx.x;
  const int nBB = 3*14*2*512, nB1 = 12*8*512, nBF = 38*6*512, nTW = 66*9;
  if (t < nBB) {
    int g = t / 14336; int r = t - g*14336;
    int chunk = r >> 10; int r2 = r & 1023;
    int nb = r2 >> 9; int lane = (r2 >> 4) & 31; int h = r2 & 15;
    int k = chunk*32 + ((lane >> 4) << 3) + (h & 7) + ((h >> 3) << 4);
    int j = nb*16 + (lane & 15);
    BBf[t] = (bf16)((k < 441 && j < 22) ? bank[(22*g + j)*441 + k] : 0.f);
    return;
  }
  t -= nBB;
  if (t < nB1) {
    int chunk = t >> 12; int r = t & 4095;
    int nb = r >> 9; int lane = (r >> 4) & 31; int h = r & 15;
    int k = chunk*32 + ((lane >> 4) << 3) + (h & 7) + ((h >> 3) << 4);
    int c = nb*16 + (lane & 15);
    B1f[t] = (bf16)((k < 363) ? pw1[c*363 + k] : 0.f);
    return;
  }
  t -= nB1;
  if (t < nBF) {
    int chunk = t / 3072; int r = t - chunk*3072;
    int nb = r >> 9; int lane = (r >> 4) & 31; int h = r & 15;
    int k = chunk*32 + ((lane >> 4) << 3) + (h & 7) + ((h >> 3) << 4);
    int oc = nb*16 + (lane & 15);
    BFf[t] = (bf16)((k < 1188 && oc < 66) ? fw[oc*1188 + k] : 0.f);
    return;
  }
  t -= nBF;
  if (t < nTW) {
    int oc = t / 9; int s = t - oc*9;
    tw[t] = tcw[oc*27 + s] + tcw[oc*27 + 9 + s] + tcw[oc*27 + 18 + s];
  }
}

__global__ void k_zero16(v4u* p, int n) {
  int t = blockIdx.x * blockDim.x + threadIdx.x;
  if (t < n) { v4u z = {0u,0u,0u,0u}; p[t] = z; }
}

// fill all padded input copies from x (one read, three writes); grid (PB256, 3, NN)
__global__ void k_fill(const float* __restrict__ x, bf16* xpg, float* xpf, bf16* xp1,
                       const int* __restrict__ pdil) {
  int p = blockIdx.x * blockDim.x + threadIdx.x;
  if (p >= PP) return;
  const int c = blockIdx.y, n = blockIdx.z;
  int dil = pdil[0]; dil = dil < 1 ? 1 : (dil > 8 ? 8 : dil);
  int y = p / WW, xx = p - (p / WW)*WW;
  float v = x[(size_t)(n*3 + c)*PP + p];
  size_t gi = ((size_t)(n*3 + c)*HG + y + 10*dil)*WG + xx + 10*dil;
  xpg[gi] = (bf16)v;
  xpf[gi] = v;
  xp1[((size_t)(n*4 + c)*H1 + y + 2)*W1 + xx + 2] = (bf16)v;
}

// ---------------- grouped inverse-gaussian bank conv (WMMA, branchless) ----------------
__global__ __launch_bounds__(256)
void k_xgauss(const bf16* __restrict__ xpg, const bf16* __restrict__ BBf,
              bf16* catp, const int* __restrict__ pdil) {
  __shared__ v16bf Af[128];
  const int tid = threadIdx.x, lane = tid & 31, w = tid >> 5;
  const int mt = blockIdx.x, g = blockIdx.y, n = blockIdx.z;
  const int p0 = mt * 64;
  int dil = pdil[0]; dil = dil < 1 ? 1 : (dil > 8 ? 8 : dil);
  const int msub = (w >> 1) * 16, nsub = (w & 1) * 16;

  const int laneS = (tid >> 1) & 31;
  const int selS  = laneS >> 4;
  const int mrow  = (tid >> 6) * 16 + (laneS & 15);
  const int kadd  = (tid & 1) << 4;
  const int pix   = p0 + mrow;                 // may run past PP: rows masked at store
  const int yP    = pix / WW, xP = pix - (pix / WW) * WW;
  const bf16* xg  = xpg + (size_t)(n*3 + g)*HG*WG + yP*WG + xP;

  const v16bf* Bg = (const v16bf*)BBf;
  v8f acc = {0.f,0.f,0.f,0.f,0.f,0.f,0.f,0.f};

  for (int chunk = 0; chunk < 14; ++chunk) {
    const int kc = chunk * 32;
    v8bf tmp;
#pragma unroll
    for (int j = 0; j < 8; ++j) {               // branchless: K-pad rows hit zero B
      int gk = kc + 8*selS + j + kadd;
      int ty = (gk * 6242) >> 17;               // /21 (valid gk<13107)
      int tx = gk - ty*21;
      tmp[j] = xg[(ty*dil)*WG + tx*dil];
    }
    ((v8bf*)Af)[tid] = tmp;
    __syncthreads();
    v16bf a = Af[(w >> 1)*32 + lane];
    v16bf b = Bg[((size_t)(g*14 + chunk)*2 + (nsub >> 4))*32 + lane];
    acc = WMMA_BF16(a, b, acc);
    __syncthreads();
  }
  const int col = lane & 15, sel = lane >> 4;
  const int ocl = nsub + col;
  if (ocl < 22) {
    bf16* dst = catp + (size_t)(n*CCH + 22*g + ocl) * PPAD;
    const int pbq = p0 + msub + 8*sel;           // 8 consecutive pixels
    const int y0 = pbq / WW, x0 = pbq - (pbq / WW)*WW;
    const int base = (y0 + 1)*PADW + (x0 + 1);
#pragma unroll
    for (int r = 0; r < 8; ++r) {
      int xr = x0 + r;                           // at most one row wrap in 8
      int addr = base + r + ((xr >= WW) ? (PADW - WW) : 0);
      if (pbq + r < PP) dst[addr] = (bf16)acc[r];
    }
  }
}

// ------------- predictor conv1 (11x11 -> 128) fused with 84x84 avg-pool -------------
__global__ __launch_bounds__(256)
void k_pred1(const bf16* __restrict__ xp1, const bf16* __restrict__ B1f,
             const float* __restrict__ pb1, float* pooled) {
  __shared__ v16bf Af[128];
  __shared__ float accs[8][2][16];
  const int tid = threadIdx.x, lane = tid & 31, w = tid >> 5;
  const int cblk = blockIdx.x, ij = blockIdx.y, n = blockIdx.z;
  const int bi = ij / 3, bj = ij % 3;
  const int msub = (w >> 1) * 16, nsub = (w & 1) * 16;
  const int col = lane & 15, sel = lane >> 4;
  ((float*)accs)[tid] = 0.f;
  __syncthreads();
  const float biasv = pb1[cblk*32 + nsub + col];

  const int laneS = (tid >> 1) & 31;
  const int selS  = laneS >> 4;
  const int mrow  = (tid >> 6) * 16 + (laneS & 15);
  const int kadd  = (tid & 1) << 4;
  const bf16* xp1n = xp1 + (size_t)n*4*H1*W1;

  const v16bf* Bg = (const v16bf*)B1f;
  const size_t nbW = (size_t)(cblk*2 + (nsub >> 4));

  for (int mt = 0; mt < 111; ++mt) {                 // 7056 pixels in 64-tiles
    v8f acc = {0.f,0.f,0.f,0.f,0.f,0.f,0.f,0.f};
    const int q = mt*64 + mrow;                      // invalid rows masked at pooling
    const int py = q / 84, px = q - py*84;
    const int obase = (bi*84 + py)*W1 + bj*84 + px;  // padded (+2,+2) folded into taps
    for (int chunk = 0; chunk < 12; ++chunk) {
      const int kc = chunk * 32;
      v8bf tmp;
#pragma unroll
      for (int j = 0; j < 8; ++j) {                  // branchless: ch3 zero, halo zero
        int gk = kc + 8*selS + j + kadd;
        int c  = (gk * 1084) >> 17;                  // /121 (valid gk<1424)
        int t2 = gk - 121*c;
        int r11 = (t2 * 11916) >> 17;                // /11
        int c11 = t2 - 11*r11;
        tmp[j] = xp1n[(size_t)c*(H1*W1) + obase + r11*W1 + c11];
      }
      ((v8bf*)Af)[tid] = tmp;
      __syncthreads();
      v16bf a = Af[(w >> 1)*32 + lane];
      v16bf b = Bg[((size_t)chunk*8 + nbW)*32 + lane];
      acc = WMMA_BF16(a, b, acc);
      __syncthreads();
    }
    float s = 0.f;                                   // bias + leaky + pooled partial
#pragma unroll
    for (int r = 0; r < 8; ++r) {
      int qq = mt*64 + msub + r + 8*sel;
      if (qq < 7056) {
        float v = acc[r] + biasv;
        s += (v > 0.f) ? v : 0.2f * v;
      }
    }
    accs[w][sel][col] += s;
    __syncthreads();
  }
  if (tid < 32) {
    int nb = tid >> 4, c2 = tid & 15;
    float s = 0.f;
    for (int ww = nb; ww < 8; ww += 2)
      s += accs[ww][0][c2] + accs[ww][1][c2];
    pooled[(size_t)(n*128 + cblk*32 + tid)*9 + ij] = s / 7056.f;
  }
}

// ---------------- predictor tail (tiny, fp32 VALU) ----------------
__global__ void k_h2(const float* __restrict__ pooled, const float* __restrict__ pw2,
                     float* h2) {
  int t = blockIdx.x * blockDim.x + threadIdx.x;
  if (t >= NN*441*9) return;
  int n = t / (441*9); int r = t % (441*9); int o = r / 9; int ij = r % 9;
  const float* pc = pooled + (size_t)n*128*9 + ij;
  const float* wv = pw2 + (size_t)o*128;
  float s = 0.f;
  for (int c = 0; c < 128; ++c) s += wv[c] * pc[c*9];
  h2[t] = (s > 0.f) ? s : 0.2f * s;
}
__global__ void k_h3(const float* __restrict__ h2, float* h3) {
  int t = blockIdx.x * blockDim.x + threadIdx.x;
  if (t >= NN*441*3) return;
  int n = t / (441*3); int r = t % (441*3); int o = r / 3; int i = r % 3;
  const float* p = h2 + (size_t)n*441*9 + o*9 + i*3;
  h3[(size_t)n*1323 + o*3 + i] = (p[0] + p[1] + p[2]) * (1.f/3.f);
}
__global__ void k_h4(const float* __restrict__ h3, const float* __restrict__ pw3,
                     float* h4) {
  int t = blockIdx.x * blockDim.x + threadIdx.x;
  if (t >= NN*441*3) return;
  int n = t / (441*3); int r = t % (441*3); int o = r / 3; int i = r % 3;
  const float* hv = h3 + (size_t)n*1323 + i;
  const float* wv = pw3 + (size_t)o*441;
  float s = 0.f;
  for (int k = 0; k < 441; ++k) s += wv[k] * hv[k*3];
  h4[(size_t)n*1323 + o*3 + i] = s;
}
__global__ __launch_bounds__(256)
void k_softmax(const float* __restrict__ h4, float* kad) {
  __shared__ float sm[441];
  __shared__ float red[256];
  const int i = blockIdx.x, n = blockIdx.y, tid = threadIdx.x;
  float mx = -1e30f;
  for (int o = tid; o < 441; o += 256) {
    float v = h4[(size_t)n*1323 + o*3 + i];
    sm[o] = v; mx = fmaxf(mx, v);
  }
  red[tid] = mx; __syncthreads();
  for (int s = 128; s > 0; s >>= 1) { if (tid < s) red[tid] = fmaxf(red[tid], red[tid+s]); __syncthreads(); }
  mx = red[0]; __syncthreads();
  float sum = 0.f;
  for (int o = tid; o < 441; o += 256) { float e = expf(sm[o] - mx); sm[o] = e; sum += e; }
  red[tid] = sum; __syncthreads();
  for (int s = 128; s > 0; s >>= 1) { if (tid < s) red[tid] += red[tid+s]; __syncthreads(); }
  float inv = 1.f / red[0];
  for (int o = tid; o < 441; o += 256) kad[(size_t)n*1323 + o*3 + i] = sm[o] * inv;
}

// ---------------- per-sample dynamic 21x21 conv (fp32, branchless); grid (PB256, NN) ---
__global__ void k_dyn(const float* __restrict__ xpf, const float* __restrict__ kad,
                      float* out1p, const int* __restrict__ pdil) {
  int p = blockIdx.x * blockDim.x + threadIdx.x;
  if (p >= PP) return;
  const int n = blockIdx.y;
  int y = p / WW, xo = p % WW;
  int dil = pdil[0]; dil = dil < 1 ? 1 : (dil > 8 ? 8 : dil);
  float s = 0.f;
  for (int c = 0; c < 3; ++c) {
    const float* xc = xpf + ((size_t)(n*3 + c)*HG + y)*WG + xo;
    const float* kn = kad + (size_t)n*1323 + c*441;
    for (int b = 0; b < 21; ++b) {
      const float* rowp = xc + b*dil*WG;
      for (int cc = 0; cc < 21; ++cc)
        s += rowp[cc*dil] * kn[b*21 + cc];
    }
  }
  out1p[(size_t)n*(PADW*PADW) + (y + 1)*PADW + (xo + 1)] = s;
}

// feat = tconv_b + conv3x3(out1); channels 66..131 of padded concat; grid (PB256, 66, NN)
__global__ void k_feat(const float* __restrict__ out1p, const float* __restrict__ tw,
                       const float* __restrict__ tcb, bf16* catp) {
  int p = blockIdx.x * blockDim.x + threadIdx.x;
  if (p >= PP) return;
  const int oc = blockIdx.y, n = blockIdx.z;
  int y = p / WW, x = p - (p / WW)*WW;
  const float* o1 = out1p + (size_t)n*(PADW*PADW) + y*PADW + x;  // halo folds the -1
  const float* wv = tw + oc*9;
  float s = tcb[oc];
#pragma unroll
  for (int ky = 0; ky < 3; ++ky)
#pragma unroll
    for (int kx = 0; kx < 3; ++kx)
      s += o1[ky*PADW + kx] * wv[ky*3 + kx];
  catp[(size_t)(n*CCH + 66 + oc)*PPAD + (y + 1)*PADW + (x + 1)] = (bf16)s;
}

// ax_output = conv3x3(x, sout_w) + sout_b (fp32, branchless); grid (PB256, 3, NN)
__global__ void k_ax(const float* __restrict__ xpf, const float* __restrict__ sw,
                     const float* __restrict__ sb, float* outax,
                     const int* __restrict__ pdil) {
  int p = blockIdx.x * blockDim.x + threadIdx.x;
  if (p >= PP) return;
  const int oc = blockIdx.y, n = blockIdx.z;
  int y = p / WW, xo = p - (p / WW)*WW;
  int dil = pdil[0]; dil = dil < 1 ? 1 : (dil > 8 ? 8 : dil);
  const int h0 = 10*dil - 1;
  float s = sb[oc];
  for (int c = 0; c < 3; ++c) {
    const float* xc = xpf + ((size_t)(n*3 + c)*HG + y + h0)*WG + xo + h0;
    const float* wv = sw + oc*27 + c*9;
#pragma unroll
    for (int ky = 0; ky < 3; ++ky)
#pragma unroll
      for (int kx = 0; kx < 3; ++kx)
        s += xc[ky*WG + kx] * wv[ky*3 + kx];
  }
  outax[(size_t)(n*3 + oc)*PP + p] = s;
}

// ---------------- fused conv (dominant GEMM, WMMA, double-buffered LDS) ----------------
// per (n, 64-pixel tile): M=64, N=96 (66 valid), K=1216 (1188 valid)
__global__ __launch_bounds__(256)
void k_fused(const bf16* __restrict__ catp, const bf16* __restrict__ BFf,
             const float* __restrict__ fb, float* outf) {
  __shared__ v16bf Af[2][128];
  const int tid = threadIdx.x, lane = tid & 31, w = tid >> 5;
  const int mt = blockIdx.x, n = blockIdx.y;
  const int p0 = mt * 64;
  const int msub = (w >> 1) * 16, nsub = (w & 1) * 16;
  const bf16* catn = catp + (size_t)n*CCH*PPAD;

  const int laneS = (tid >> 1) & 31;
  const int selS  = laneS >> 4;
  const int mrow  = (tid >> 6) * 16 + (laneS & 15);
  const int kadd  = (tid & 1) << 4;
  const int pix   = p0 + mrow;                 // may run past PP into zero rows
  const int yP    = pix / WW, xP = pix - (pix / WW) * WW;
  const bf16* src = catn + yP*PADW + xP;       // (+1,+1) halo folded via taps

  const v16bf* Bg = (const v16bf*)BFf;
  v8f zero = {0.f,0.f,0.f,0.f,0.f,0.f,0.f,0.f};
  v8f acc[3]; acc[0] = zero; acc[1] = zero; acc[2] = zero;

  auto stage = [&](int chunk, int buf) {
    const int kc = chunk * 32;
    v8bf tmp;
#pragma unroll
    for (int j = 0; j < 8; ++j) {               // fully branchless im2col
      int gk = kc + 8*selS + j + kadd;          // gk<1216 -> c<=135 (zero channels)
      int c   = (gk * 14564) >> 17;             // /9
      int t2  = gk - 9*c;
      int t2d = (t2 * 43691) >> 17;             // /3
      int t2m = t2 - 3*t2d;
      tmp[j] = src[(size_t)c*PPAD + t2d*PADW + t2m];
    }
    ((v8bf*)(&Af[buf][0]))[tid] = tmp;
  };

  stage(0, 0);
  __syncthreads();
  for (int chunk = 0; chunk < 38; ++chunk) {
    const int buf = chunk & 1;
    if (chunk + 1 < 38)
      __builtin_prefetch(Bg + ((size_t)(chunk + 1)*6)*32 + lane, 0, 1);
    v16bf a = Af[buf][(w >> 1)*32 + lane];
#pragma unroll
    for (int nb = 0; nb < 3; ++nb) {
      v16bf b = Bg[((size_t)chunk*6 + nb*2 + (nsub >> 4))*32 + lane];
      acc[nb] = WMMA_BF16(a, b, acc[nb]);
    }
    if (chunk + 1 < 38) stage(chunk + 1, buf ^ 1);
    __syncthreads();
  }
  const int col = lane & 15, sel = lane >> 4;
  const int pbq = p0 + msub + 8*sel;             // 8 consecutive pixels per lane
#pragma unroll
  for (int nb = 0; nb < 3; ++nb) {
    int oc = nb*32 + nsub + col;
    if (oc < 66) {
      float bias = fb[oc];
      float* dst = outf + (size_t)(n*66 + oc)*PP + pbq;
#pragma unroll
      for (int r = 0; r < 8; ++r)
        if (pbq + r < PP) dst[r] = acc[nb][r] + bias;
    }
  }
}

// ---------------- launcher ----------------
extern "C" void kernel_launch(void* const* d_in, const int* in_sizes, int n_in,
                              void* d_out, int out_size, void* d_ws, size_t ws_size,
                              hipStream_t stream) {
  (void)in_sizes; (void)n_in; (void)out_size; (void)ws_size;
  const float* x    = (const float*)d_in[0];
  const float* bank = (const float*)d_in[1];
  const float* pw1  = (const float*)d_in[2];
  const float* pb1  = (const float*)d_in[3];
  const float* pw2  = (const float*)d_in[4];
  const float* pw3  = (const float*)d_in[5];
  const float* tcw  = (const float*)d_in[6];
  const float* tcb  = (const float*)d_in[7];
  const float* fw   = (const float*)d_in[8];
  const float* fb   = (const float*)d_in[9];
  const float* sw   = (const float*)d_in[10];
  const float* sb   = (const float*)d_in[11];
  const int*   dil  = (const int*)d_in[12];

  char* ws = (char*)d_ws;
  size_t off = 0;
  auto alloc = [&](size_t bytes) { size_t o = off; off += (bytes + 255) & ~(size_t)255; return o; };
  bf16*  CATP  = (bf16*)(ws + alloc((size_t)NN*CCH*PPAD*2));     // padded concat
  bf16*  XPG   = (bf16*)(ws + alloc((size_t)NN*3*HG*WG*2));      // x bf16, halo 10*dil
  float* XPF   = (float*)(ws + alloc((size_t)NN*3*HG*WG*4));     // x f32, halo 10*dil
  bf16*  XP1   = (bf16*)(ws + alloc((size_t)NN*4*H1*W1*2));      // x bf16, halo 2, 4ch
  float* OUT1P = (float*)(ws + alloc((size_t)NN*PADW*PADW*4));   // dyn conv out, halo 1
  bf16*  BBf   = (bf16*)(ws + alloc((size_t)3*14*2*512*2));
  bf16*  B1f   = (bf16*)(ws + alloc((size_t)12*8*512*2));
  bf16*  BFf   = (bf16*)(ws + alloc((size_t)38*6*512*2));
  float* TW    = (float*)(ws + alloc((size_t)66*9*4));
  float* POOL  = (float*)(ws + alloc((size_t)NN*128*9*4));
  float* H2    = (float*)(ws + alloc((size_t)NN*441*9*4));
  float* H3    = (float*)(ws + alloc((size_t)NN*441*3*4));
  float* H4    = (float*)(ws + alloc((size_t)NN*441*3*4));
  float* KAD   = (float*)(ws + alloc((size_t)NN*1323*4));

  float* outFused = (float*)d_out;                       // (8,66,258,258)
  float* outAx    = outFused + (size_t)NN*66*PP;         // (8,3,258,258)

  // zero padded buffers (16B vector stores)
  const int zCat = NN*CCH*PPAD*2/16, zXpg = NN*3*HG*WG*2/16, zXpf = NN*3*HG*WG*4/16;
  const int zXp1 = NN*4*H1*W1*2/16, zO1 = NN*PADW*PADW*4/16;
  k_zero16<<<(zCat + 255)/256, 256, 0, stream>>>((v4u*)CATP, zCat);
  k_zero16<<<(zXpg + 255)/256, 256, 0, stream>>>((v4u*)XPG, zXpg);
  k_zero16<<<(zXpf + 255)/256, 256, 0, stream>>>((v4u*)XPF, zXpf);
  k_zero16<<<(zXp1 + 255)/256, 256, 0, stream>>>((v4u*)XP1, zXp1);
  k_zero16<<<(zO1  + 255)/256, 256, 0, stream>>>((v4u*)OUT1P, zO1);

  const int nPrep = 3*14*2*512 + 12*8*512 + 38*6*512 + 66*9;
  k_prep<<<(nPrep + 255)/256, 256, 0, stream>>>(bank, pw1, fw, tcw, BBf, B1f, BFf, TW);
  k_fill<<<dim3(PB256, 3, NN), 256, 0, stream>>>(x, XPG, XPF, XP1, dil);

  k_xgauss<<<dim3(MT64, 3, NN), 256, 0, stream>>>(XPG, BBf, CATP, dil);
  k_pred1 <<<dim3(4, 9, NN),    256, 0, stream>>>(XP1, B1f, pb1, POOL);

  k_h2<<<(NN*441*9 + 255)/256, 256, 0, stream>>>(POOL, pw2, H2);
  k_h3<<<(NN*441*3 + 255)/256, 256, 0, stream>>>(H2, H3);
  k_h4<<<(NN*441*3 + 255)/256, 256, 0, stream>>>(H3, pw3, H4);
  k_softmax<<<dim3(3, NN), 256, 0, stream>>>(H4, KAD);

  k_dyn <<<dim3(PB256, NN), 256, 0, stream>>>(XPF, KAD, OUT1P, dil);
  k_feat<<<dim3(PB256, 66, NN), 256, 0, stream>>>(OUT1P, TW, tcb, CATP);
  k_ax  <<<dim3(PB256, 3, NN), 256, 0, stream>>>(XPF, sw, sb, outAx, dil);

  k_fused<<<dim3(MT64, NN), 256, 0, stream>>>(CATP, BFf, fb, outFused);
}
